// AdaptiveModel_35974646072131
// MI455X (gfx1250) — matrix operand order repaired and checked by reading, not compiled, not run on hardware
//
#include <hip/hip_runtime.h>
#include <hip/hip_bf16.h>
#include <stdint.h>

// ---------------------------------------------------------------------------
// Adaptive-attention LSTM captioner for MI455X (gfx1250).
// All dense GEMMs run through a bf16 WMMA (v_wmma_f32_16x16x32_bf16) tile
// kernel with f32 accumulation: 128x64 macro-tile, 8 waves, each wave owns a
// 32x32 slab = 4 WMMAs per 32-deep k-step (2 A-frags x 2 B-frags), branchless
// clamped staging, software-pipelined global loads.
// V@Wv is hoisted out of the 30-step scan (loop-invariant).
// ---------------------------------------------------------------------------

typedef __bf16 bf16;
typedef __attribute__((ext_vector_type(16))) __bf16 bf16x16;
typedef __attribute__((ext_vector_type(8)))  float  f32x8;

#define NB    128     // batch
#define NK    64      // image patches
#define ND    1536    // image feat dim
#define NEM   512     // embedding dim
#define NH    1024    // hidden
#define NA    1024    // attention dim
#define NVOC  10000
#define NT    30      // MAX_LEN
#define NTL   31      // MAX_LEN+1

__device__ __forceinline__ bf16 f2bf(float f) {
  union { float f; uint32_t u; } x; x.f = f;
  uint32_t r = x.u + 0x7FFFu + ((x.u >> 16) & 1u);   // round-to-nearest-even
  union { unsigned short s; bf16 b; } y; y.s = (unsigned short)(r >> 16);
  return y.b;
}
__device__ __forceinline__ float sigm(float x) { return 1.0f / (1.0f + __expf(-x)); }

union FragU { uint4 q[2]; bf16x16 v; };
union BU    { uint4 q; bf16 h[8]; };

// ---------------------------------------------------------------------------
// bf16 WMMA GEMM: C[M,N] (f32 and/or bf16) = act(A[M,K] @ B[K,N] + bias)
//  - A row-major bf16, leading dim lda, optional row gather:
//      a_row = amap[row / rpb] * rpb + row % rpb (folds argsort permutation)
//  - B row-major bf16 [K,N]
//  - block = 256 threads (8 waves); macro tile 128(M) x 64(N); each wave owns
//    a 32x32 slab: 2 A-fragments x 2 B-fragments -> 4 WMMAs per k-step.
//  - K must be a multiple of 32. M/N edges handled by address clamping
//    (garbage rows/cols only feed never-stored outputs) -> branchless staging.
// ---------------------------------------------------------------------------
__global__ __launch_bounds__(256) void gemm_bf16_wmma(
    const bf16* __restrict__ A, int lda, const bf16* __restrict__ Bm,
    const float* __restrict__ bias, float* __restrict__ C, bf16* __restrict__ Cbf,
    const int* __restrict__ amap, int rpb, int M, int N, int K, int relu) {
  __shared__ bf16 As[128][32];  // [m][k] row-major        (8 KB)
  __shared__ bf16 Bt[64][32];   // [n][k] transposed stage (4 KB)

  const int tid = threadIdx.x;
  const int n0  = blockIdx.x * 64;
  const int m0  = blockIdx.y * 128;
  const int wv  = tid >> 5, lane = tid & 31;
  const int waveM = wv >> 1;          // 0..3 -> m offset 32*waveM
  const int waveN = wv & 1;           // 0..1 -> n offset 32*waveN
  const int lrow  = lane & 15, lg = lane >> 4;

  // staging indices (256 threads)
  const int mrow = tid >> 1;          // 0..127  A m-row
  const int kc   = (tid & 1) * 16;    // 0 or 16 A k-col (16 elements each)
  const int krow = tid & 31;          // 0..31   B k-row
  const int nc   = (tid >> 5) * 8;    // 0,8,..,56 B n-col
  const bool fastN = (n0 + 64 <= N);  // uniform per block

  // branchless clamped A base (edge rows feed never-stored outputs)
  int grow = m0 + mrow; if (grow > M - 1) grow = M - 1;
  const int arow = amap ? (amap[grow / rpb] * rpb + (grow % rpb)) : grow;
  const size_t abase = (size_t)arow * (size_t)lda;

  uint4 av0, av1, bvreg;
  auto load_tile = [&](int k0) {
    av0 = *(const uint4*)(A + abase + (size_t)(k0 + kc));
    av1 = *(const uint4*)(A + abase + (size_t)(k0 + kc + 8));
    const size_t brow = (size_t)(k0 + krow) * (size_t)N;
    if (fastN) {
      bvreg = *(const uint4*)(Bm + brow + (size_t)(n0 + nc));
    } else {
      BU bu;
#pragma unroll
      for (int j = 0; j < 8; ++j) {
        int gn = n0 + nc + j; if (gn > N - 1) gn = N - 1;   // clamp, no zero-fill
        bu.h[j] = Bm[brow + gn];
      }
      bvreg = bu.q;
    }
    if (k0 + 32 < K) {
      __builtin_prefetch(A + abase + (size_t)(k0 + 32 + kc), 0, 1);
      __builtin_prefetch(Bm + (size_t)(k0 + 32 + krow) * (size_t)N + (size_t)(n0 + nc), 0, 1);
    }
  };

  f32x8 acc00 = {0.f,0.f,0.f,0.f,0.f,0.f,0.f,0.f};
  f32x8 acc01 = acc00, acc10 = acc00, acc11 = acc00;

  load_tile(0);
  for (int k0 = 0; k0 < K; k0 += 32) {
    // commit staged registers to LDS
    *(uint4*)&As[mrow][kc]     = av0;
    *(uint4*)&As[mrow][kc + 8] = av1;
    {
      BU bu; bu.q = bvreg;
#pragma unroll
      for (int j = 0; j < 8; ++j) Bt[nc + j][krow] = bu.h[j];
    }
    __syncthreads();
    // issue next tile's global loads so they overlap the WMMAs below
    if (k0 + 32 < K) load_tile(k0 + 32);

    // A fragments (16x32): lane = M row, K groups of 8 selected by lg
    FragU fa0, fa1, fb0, fb1;
    const uint4* arp0 = (const uint4*)&As[waveM * 32 + lrow][0];
    fa0.q[0] = arp0[lg];     fa0.q[1] = arp0[2 + lg];
    const uint4* arp1 = (const uint4*)&As[waveM * 32 + 16 + lrow][0];
    fa1.q[0] = arp1[lg];     fa1.q[1] = arp1[2 + lg];
    // B fragments (32x16): lane = N col, K groups of 8 selected by lg (matches A)
    const uint4* brp0 = (const uint4*)&Bt[waveN * 32 + lrow][0];
    fb0.q[0] = brp0[lg];     fb0.q[1] = brp0[2 + lg];
    const uint4* brp1 = (const uint4*)&Bt[waveN * 32 + 16 + lrow][0];
    fb1.q[0] = brp1[lg];     fb1.q[1] = brp1[2 + lg];

    acc00 = __builtin_amdgcn_wmma_f32_16x16x32_bf16(
        false, fa0.v, false, fb0.v, (short)0, acc00, false, false);
    acc01 = __builtin_amdgcn_wmma_f32_16x16x32_bf16(
        false, fa0.v, false, fb1.v, (short)0, acc01, false, false);
    acc10 = __builtin_amdgcn_wmma_f32_16x16x32_bf16(
        false, fa1.v, false, fb0.v, (short)0, acc10, false, false);
    acc11 = __builtin_amdgcn_wmma_f32_16x16x32_bf16(
        false, fa1.v, false, fb1.v, (short)0, acc11, false, false);
    __syncthreads();
  }

  // store 4 tiles
  const int ncol0 = n0 + waveN * 32 + lrow;
  const int mb0   = m0 + waveM * 32 + lg * 8;
  auto store_tile = [&](const f32x8& acc, int mb, int ncol) {
    if (ncol < N) {
      const float bvv = bias ? bias[ncol] : 0.0f;
#pragma unroll
      for (int v = 0; v < 8; ++v) {
        int m = mb + v;
        if (m < M) {
          float r = acc[v] + bvv;
          if (relu) r = fmaxf(r, 0.0f);
          size_t idx = (size_t)m * (size_t)N + ncol;
          if (C)   C[idx]   = r;
          if (Cbf) Cbf[idx] = f2bf(r);
        }
      }
    }
  };
  store_tile(acc00, mb0,      ncol0);
  store_tile(acc01, mb0,      ncol0 + 16);
  store_tile(acc10, mb0 + 16, ncol0);
  store_tile(acc11, mb0 + 16, ncol0 + 16);
}

// ---------------------------------------------------------------------------
// Stable descending rank-sort by caption length (B=128, one block).
// ---------------------------------------------------------------------------
__global__ __launch_bounds__(128) void sort_prep(
    const int* __restrict__ cap, const int* __restrict__ w_in,
    int* __restrict__ order, int* __restrict__ w_s, int* __restrict__ lens_s,
    float* __restrict__ out_target, float* __restrict__ out_lens) {
  __shared__ int sl[NB];
  __shared__ int sord[NB];
  const int i = threadIdx.x;
  sl[i] = cap[i];
  __syncthreads();
  int my = sl[i], rank = 0;
  for (int j = 0; j < NB; ++j) {
    int lj = sl[j];
    if (lj > my || (lj == my && j < i)) rank++;
  }
  sord[rank] = i;
  __syncthreads();
  const int src = sord[i];
  order[i] = src;
  const int L = cap[src] - 1;
  lens_s[i]   = L;
  out_lens[i] = (float)L;
  for (int t = 0; t < NTL; ++t) w_s[i * NTL + t] = w_in[src * NTL + t];
  for (int t = 0; t < NT;  ++t) out_target[i * NT + t] = (float)w_in[src * NTL + t + 1];
}

__global__ __launch_bounds__(256) void cvt_bf16(
    const float* __restrict__ src, bf16* __restrict__ dst, int n) {
  int i = blockIdx.x * 256 + threadIdx.x;
  if (i < n) dst[i] = f2bf(src[i]);
}

__global__ __launch_bounds__(256) void add2(
    const float* __restrict__ a, const float* __restrict__ b,
    float* __restrict__ o, int n) {
  int i = blockIdx.x * 256 + threadIdx.x;
  if (i < n) o[i] = a[i] + b[i];
}

__global__ __launch_bounds__(256) void mean_im(
    const float* __restrict__ im, bf16* __restrict__ out) {
  int idx = blockIdx.x * 256 + threadIdx.x;
  if (idx >= NB * ND) return;
  int b = idx / ND, d = idx % ND;
  const float* p = im + (size_t)b * NK * ND + d;
  float sm = 0.f;
#pragma unroll 8
  for (int k = 0; k < NK; ++k) sm += p[(size_t)k * ND];
  out[idx] = f2bf(sm * (1.0f / (float)NK));
}

__global__ __launch_bounds__(256) void init_state(
    float* __restrict__ h, float* __restrict__ c, bf16* __restrict__ xh) {
  int idx = blockIdx.x * 256 + threadIdx.x;
  if (idx >= NB * NH) return;
  h[idx] = 0.f; c[idx] = 0.f;
  int b = idx >> 10, j = idx & 1023;
  xh[(size_t)b * 2048 + 1024 + j] = f2bf(0.f);
}

// xh[:, :512] = emb[token], xh[:, 512:1024] = g  (h half maintained elsewhere)
__global__ __launch_bounds__(256) void build_x(
    const bf16* __restrict__ emb, const bf16* __restrict__ g,
    const int* __restrict__ w_s, bf16* __restrict__ xh, int t) {
  int idx = blockIdx.x * 256 + threadIdx.x;
  if (idx >= NB * 1024) return;
  int b = idx >> 10, j = idx & 1023;
  bf16 v;
  if (j < NEM) v = emb[(size_t)w_s[b * NTL + t] * NEM + j];
  else         v = g[(size_t)b * NEM + (j - NEM)];
  xh[(size_t)b * 2048 + j] = v;
}

__global__ __launch_bounds__(256) void lstm_pointwise(
    const float* __restrict__ gates, const float* __restrict__ spre,
    float* __restrict__ c, float* __restrict__ h, float* __restrict__ s,
    bf16* __restrict__ xh, bf16* __restrict__ sbf) {
  int idx = blockIdx.x * 256 + threadIdx.x;
  if (idx >= NB * NH) return;
  int b = idx >> 10, j = idx & 1023;
  const float* gr = gates + (size_t)b * (4 * NH);
  float ig = sigm(gr[j]);
  float fg = sigm(gr[NH + j]);
  float gg = tanhf(gr[2 * NH + j]);
  float og = sigm(gr[3 * NH + j]);
  float cn = fg * c[idx] + ig * gg;
  float tc = tanhf(cn);
  float hn = og * tc;
  float sv = sigm(spre[idx]) * tc;
  c[idx] = cn; h[idx] = hn; s[idx] = sv;
  xh[(size_t)b * 2048 + 1024 + j] = f2bf(hn);
  sbf[idx] = f2bf(sv);
}

// out[r] = sum_a tanh(X[r,a] + hp[r/hpDiv, a]) * wa[a] + ba
__global__ __launch_bounds__(256) void attn_dot(
    const float* __restrict__ X, const float* __restrict__ hp, int hpDiv,
    const float* __restrict__ wa, const float* __restrict__ ba,
    float* __restrict__ out) {
  const int r = blockIdx.x, tid = threadIdx.x;
  const float* x  = X  + (size_t)r * NA;
  const float* hr = hp + (size_t)(r / hpDiv) * NA;
  __shared__ float red[256];
  float p = 0.f;
  for (int a = tid; a < NA; a += 256) p += tanhf(x[a] + hr[a]) * wa[a];
  red[tid] = p; __syncthreads();
  for (int st = 128; st > 0; st >>= 1) {
    if (tid < st) red[tid] += red[tid + st];
    __syncthreads();
  }
  if (tid == 0) out[r] = red[0] + ba[0];
}

// softmax over [zv(64), zs(1)], ctx = alpha[:64] . V, z = beta*s+(1-beta)*ctx+h
__global__ __launch_bounds__(256) void attn_softmax_ctx(
    const float* __restrict__ zv, const float* __restrict__ zs,
    const float* __restrict__ V, const float* __restrict__ s,
    const float* __restrict__ h, bf16* __restrict__ zbf) {
  const int b = blockIdx.x, tid = threadIdx.x;
  __shared__ float al[65];
  if (tid < 64) al[tid] = zv[b * NK + tid];
  if (tid == 64) al[64] = zs[b];
  __syncthreads();
  if (tid == 0) {
    float mx = al[0];
    for (int i = 1; i < 65; ++i) mx = fmaxf(mx, al[i]);
    float sm = 0.f;
    for (int i = 0; i < 65; ++i) { al[i] = __expf(al[i] - mx); sm += al[i]; }
    float inv = 1.f / sm;
    for (int i = 0; i < 65; ++i) al[i] *= inv;
  }
  __syncthreads();
  const float beta = al[64];
  const float* vb = V + (size_t)b * NK * NH;
  for (int j = tid; j < NH; j += 256) {
    float cacc = 0.f;
#pragma unroll 8
    for (int k = 0; k < NK; ++k) cacc += al[k] * vb[(size_t)k * NH + j];
    size_t idx = (size_t)b * NH + j;
    float z = beta * s[idx] + (1.0f - beta) * cacc + h[idx];
    zbf[idx] = f2bf(z);
  }
}

// masked vocab softmax -> predictions[b, t, :]
__global__ __launch_bounds__(256) void softmax_out(
    const float* __restrict__ logits, const int* __restrict__ lens_s,
    float* __restrict__ out_pred, int t) {
  const int b = blockIdx.x, tid = threadIdx.x;
  const float* row = logits + (size_t)b * NVOC;
  float* o = out_pred + ((size_t)b * NT + t) * NVOC;
  const bool on = (t < lens_s[b]);
  __shared__ float red[256];
  float mx = -3.4e38f;
  for (int v = tid; v < NVOC; v += 256) mx = fmaxf(mx, row[v]);
  red[tid] = mx; __syncthreads();
  for (int st = 128; st > 0; st >>= 1) {
    if (tid < st) red[tid] = fmaxf(red[tid], red[tid + st]);
    __syncthreads();
  }
  mx = red[0]; __syncthreads();
  float sm = 0.f;
  for (int v = tid; v < NVOC; v += 256) sm += __expf(row[v] - mx);
  red[tid] = sm; __syncthreads();
  for (int st = 128; st > 0; st >>= 1) {
    if (tid < st) red[tid] += red[tid + st];
    __syncthreads();
  }
  const float inv = 1.f / red[0];
  for (int v = tid; v < NVOC; v += 256)
    o[v] = on ? __expf(row[v] - mx) * inv : 0.f;
}

// ---------------------------------------------------------------------------
extern "C" void kernel_launch(void* const* d_in, const int* in_sizes, int n_in,
                              void* d_out, int out_size, void* d_ws, size_t ws_size,
                              hipStream_t stream) {
  (void)in_sizes; (void)n_in; (void)out_size; (void)ws_size;

  const float* im     = (const float*)d_in[0];
  const int*   w_in   = (const int*)  d_in[1];
  const int*   cap    = (const int*)  d_in[2];
  const float* emb    = (const float*)d_in[3];
  const float* W_enc  = (const float*)d_in[4];
  const float* b_enc  = (const float*)d_in[5];
  const float* W_glob = (const float*)d_in[6];
  const float* b_glob = (const float*)d_in[7];
  const float* W_ih   = (const float*)d_in[8];
  const float* b_ih   = (const float*)d_in[9];
  const float* W_hh   = (const float*)d_in[10];
  const float* b_hh   = (const float*)d_in[11];
  const float* W_xg   = (const float*)d_in[12];
  const float* b_xg   = (const float*)d_in[13];
  const float* W_hg   = (const float*)d_in[14];
  const float* b_hg   = (const float*)d_in[15];
  const float* Wv     = (const float*)d_in[16];
  const float* bv     = (const float*)d_in[17];
  const float* Wsm    = (const float*)d_in[18];
  const float* bs     = (const float*)d_in[19];
  const float* Wh     = (const float*)d_in[20];
  const float* bh     = (const float*)d_in[21];
  const float* wa     = (const float*)d_in[22];
  const float* ba     = (const float*)d_in[23];
  const float* W_out  = (const float*)d_in[24];
  const float* b_out  = (const float*)d_in[25];

  char* wsp = (char*)d_ws;
  size_t off = 0;
  auto alloc = [&](size_t bytes) -> void* {
    void* p = (void*)(wsp + off);
    off += (bytes + 255) & ~(size_t)255;
    return p;
  };

  bf16* im_bf    = (bf16*)alloc((size_t)NB * NK * ND * 2);
  bf16* emb_bf   = (bf16*)alloc((size_t)NVOC * NEM * 2);
  bf16* Wenc_bf  = (bf16*)alloc((size_t)ND * NH * 2);
  bf16* Wglob_bf = (bf16*)alloc((size_t)ND * NEM * 2);
  bf16* Wihhh_bf = (bf16*)alloc((size_t)2048 * 4096 * 2);   // [W_ih; W_hh]
  bf16* Wxhg_bf  = (bf16*)alloc((size_t)2048 * 1024 * 2);   // [W_xg; W_hg]
  bf16* Wv_bf    = (bf16*)alloc((size_t)NH * NA * 2);
  bf16* Ws_bf    = (bf16*)alloc((size_t)NH * NA * 2);
  bf16* Wh_bf    = (bf16*)alloc((size_t)NH * NA * 2);
  bf16* Wout_bf  = (bf16*)alloc((size_t)NH * NVOC * 2);
  bf16* mean_bf  = (bf16*)alloc((size_t)NB * ND * 2);
  float* Vf      = (float*)alloc((size_t)NB * NK * NH * 4);   // sorted V (f32)
  bf16* V_bf     = (bf16*)alloc((size_t)NB * NK * NH * 2);
  float* VW      = (float*)alloc((size_t)NB * NK * NA * 4);   // V@Wv + bv (hoisted)
  bf16* g_bf     = (bf16*)alloc((size_t)NB * NEM * 2);
  bf16* xh       = (bf16*)alloc((size_t)NB * 2048 * 2);       // [x | h] bf16
  float* gates   = (float*)alloc((size_t)NB * 4096 * 4);
  float* spre    = (float*)alloc((size_t)NB * NH * 4);
  float* spre2   = (float*)alloc((size_t)NB * NA * 4);
  float* hp      = (float*)alloc((size_t)NB * NA * 4);
  float* hbuf    = (float*)alloc((size_t)NB * NH * 4);
  float* cbuf    = (float*)alloc((size_t)NB * NH * 4);
  float* sbuf    = (float*)alloc((size_t)NB * NH * 4);
  bf16* sbf      = (bf16*)alloc((size_t)NB * NH * 2);
  bf16* zbf      = (bf16*)alloc((size_t)NB * NH * 2);
  float* logits  = (float*)alloc((size_t)NB * NVOC * 4);
  float* zvbuf   = (float*)alloc((size_t)NB * NK * 4);
  float* zsbuf   = (float*)alloc((size_t)NB * 4);
  float* bihhh   = (float*)alloc((size_t)4096 * 4);
  float* bxhg    = (float*)alloc((size_t)1024 * 4);
  int*  order    = (int*)alloc((size_t)NB * 4);
  int*  w_s      = (int*)alloc((size_t)NB * NTL * 4);
  int*  lens_s   = (int*)alloc((size_t)NB * 4);

  float* out_pred   = (float*)d_out;
  float* out_target = out_pred + (size_t)NB * NT * NVOC;
  float* out_lens   = out_target + (size_t)NB * NT;

  dim3 blk(256);
  auto g1 = [](int n) { return dim3((unsigned)((n + 255) / 256)); };
  auto gemm = [&](const bf16* A, int lda, const bf16* Bm, const float* bias,
                  float* C, bf16* Cbf, const int* amap, int rpb,
                  int M, int N, int K, int relu) {
    dim3 gr((unsigned)((N + 63) / 64), (unsigned)((M + 127) / 128));
    gemm_bf16_wmma<<<gr, 256, 0, stream>>>(A, lda, Bm, bias, C, Cbf, amap, rpb,
                                           M, N, K, relu);
  };

  // ---- setup --------------------------------------------------------------
  sort_prep<<<1, 128, 0, stream>>>(cap, w_in, order, w_s, lens_s, out_target, out_lens);
  cvt_bf16<<<g1(NB * NK * ND), blk, 0, stream>>>(im, im_bf, NB * NK * ND);
  cvt_bf16<<<g1(NVOC * NEM), blk, 0, stream>>>(emb, emb_bf, NVOC * NEM);
  cvt_bf16<<<g1(ND * NH), blk, 0, stream>>>(W_enc, Wenc_bf, ND * NH);
  cvt_bf16<<<g1(ND * NEM), blk, 0, stream>>>(W_glob, Wglob_bf, ND * NEM);
  cvt_bf16<<<g1(1024 * 4096), blk, 0, stream>>>(W_ih, Wihhh_bf, 1024 * 4096);
  cvt_bf16<<<g1(1024 * 4096), blk, 0, stream>>>(W_hh, Wihhh_bf + (size_t)1024 * 4096, 1024 * 4096);
  cvt_bf16<<<g1(1024 * 1024), blk, 0, stream>>>(W_xg, Wxhg_bf, 1024 * 1024);
  cvt_bf16<<<g1(1024 * 1024), blk, 0, stream>>>(W_hg, Wxhg_bf + (size_t)1024 * 1024, 1024 * 1024);
  cvt_bf16<<<g1(NH * NA), blk, 0, stream>>>(Wv, Wv_bf, NH * NA);
  cvt_bf16<<<g1(NH * NA), blk, 0, stream>>>(Wsm, Ws_bf, NH * NA);
  cvt_bf16<<<g1(NH * NA), blk, 0, stream>>>(Wh, Wh_bf, NH * NA);
  cvt_bf16<<<g1(NH * NVOC), blk, 0, stream>>>(W_out, Wout_bf, NH * NVOC);
  mean_im<<<g1(NB * ND), blk, 0, stream>>>(im, mean_bf);
  add2<<<g1(4096), blk, 0, stream>>>(b_ih, b_hh, bihhh, 4096);
  add2<<<g1(1024), blk, 0, stream>>>(b_xg, b_hg, bxhg, 1024);
  init_state<<<g1(NB * NH), blk, 0, stream>>>(hbuf, cbuf, xh);

  // V = relu(im @ W_enc + b_enc), rows gathered through `order` (sorted V)
  gemm(im_bf, ND, Wenc_bf, b_enc, Vf, V_bf, order, NK, NB * NK, NH, ND, 1);
  // g = relu(mean(im) @ W_glob + b_glob), sorted
  gemm(mean_bf, ND, Wglob_bf, b_glob, nullptr, g_bf, order, 1, NB, NEM, ND, 1);
  // Hoisted loop-invariant: VW = V @ Wv + bv
  gemm(V_bf, NH, Wv_bf, bv, VW, nullptr, nullptr, 1, NB * NK, NA, NH, 0);

  // ---- 30-step LSTM + adaptive attention ----------------------------------
  for (int t = 0; t < NT; ++t) {
    build_x<<<g1(NB * 1024), blk, 0, stream>>>(emb_bf, g_bf, w_s, xh, t);
    // gates = [x,h] @ [W_ih; W_hh] + (b_ih+b_hh)
    gemm(xh, 2048, Wihhh_bf, bihhh, gates, nullptr, nullptr, 1, NB, 4096, 2048, 0);
    // spre  = [x,h] @ [W_xg; W_hg] + (b_xg+b_hg)
    gemm(xh, 2048, Wxhg_bf, bxhg, spre, nullptr, nullptr, 1, NB, 1024, 2048, 0);
    lstm_pointwise<<<g1(NB * NH), blk, 0, stream>>>(gates, spre, cbuf, hbuf, sbuf, xh, sbf);
    // hp = h_new @ Wh + bh  (A = h half of xh, lda=2048)
    gemm(xh + 1024, 2048, Wh_bf, bh, hp, nullptr, nullptr, 1, NB, NA, NH, 0);
    // spre2 = s @ Ws + bs
    gemm(sbf, NH, Ws_bf, bs, spre2, nullptr, nullptr, 1, NB, NA, NH, 0);
    attn_dot<<<NB * NK, 256, 0, stream>>>(VW, hp, NK, wa, ba, zvbuf);
    attn_dot<<<NB, 256, 0, stream>>>(spre2, hp, 1, wa, ba, zsbuf);
    attn_softmax_ctx<<<NB, 256, 0, stream>>>(zvbuf, zsbuf, Vf, sbuf, hbuf, zbf);
    // logits = z_t @ W_out + b_out
    gemm(zbf, NH, Wout_bf, b_out, logits, nullptr, nullptr, 1, NB, NVOC, NH, 0);
    softmax_out<<<NB, 256, 0, stream>>>(logits, lens_s, out_pred, t);
  }
}
